// PyIConv_4355096838686
// MI455X (gfx1250) — matrix operand order repaired and checked by reading, not compiled, not run on hardware
//
#include <hip/hip_runtime.h>

// Problem constants (from reference): B=4, L=10000, K=16, I=256, O=256
#define B_ 4
#define L_ 10000
#define K_ 16
#define I_ 256
#define O_ 256
#define STR 264   // padded A-row stride in halfs (528B = 132 dwords ≡ 4 mod 64 banks)

typedef __attribute__((ext_vector_type(16))) _Float16 v16h;
typedef __attribute__((ext_vector_type(8)))  _Float16 v8h;
typedef __attribute__((ext_vector_type(8)))  float    v8f;

// ---------------------------------------------------------------------------
// Prologue: mask[k][i][o] (f32) -> maskT[k][o][i] (f16), tiled transpose via
// LDS so both the global read (along o) and global write (along i) coalesce.
// ---------------------------------------------------------------------------
__global__ void transpose_mask_kernel(const float* __restrict__ mask,
                                      _Float16* __restrict__ maskT) {
    __shared__ _Float16 tile[32][33];
    const int k  = blockIdx.z;
    const int i0 = blockIdx.y * 32;
    const int o0 = blockIdx.x * 32;
    const int tx = threadIdx.x;   // 0..31
    const int ty = threadIdx.y;   // 0..7
#pragma unroll
    for (int r = 0; r < 32; r += 8) {
        tile[ty + r][tx] =
            (_Float16)mask[((size_t)k * I_ + (i0 + ty + r)) * O_ + (o0 + tx)];
    }
    __syncthreads();
#pragma unroll
    for (int r = 0; r < 32; r += 8) {
        maskT[((size_t)k * O_ + (o0 + ty + r)) * I_ + (i0 + tx)] = tile[tx][ty + r];
    }
}

// ---------------------------------------------------------------------------
// Main gathered-GEMM kernel.
// Block = 128 threads (4 x wave32), computes a 32-row x 256-col output tile.
// Each wave owns 64 output columns (4 WMMA N-tiles) x both 16-row M-tiles,
// so every B fragment feeds 2 WMMAs (halves global traffic per matrix op).
// Per k (16 mask slices): cooperatively gather the 32 indexed x rows into a
// f16 LDS tile, then 8 K-chunks of 32 -> v_wmma_f32_16x16x32_f16.
// ---------------------------------------------------------------------------
__global__ __launch_bounds__(128)
void iconv_wmma_kernel(const float*      __restrict__ x,
                       const long long*  __restrict__ idx,
                       const _Float16*   __restrict__ maskT,
                       const float*      __restrict__ bias,
                       float*            __restrict__ out) {
    __shared__ _Float16 As[32 * STR];

    const int tid  = threadIdx.x;
    const int wave = tid >> 5;
    const int lane = tid & 31;

    const int m0 = blockIdx.x * 32;   // flattened (b,l) row-tile base

    // ---- loader role: 4 threads per gathered row, 64 floats each ----
    const int row  = tid >> 2;        // 0..31
    const int sub  = tid & 3;         // 0..3
    const int gm   = m0 + row;        // may straddle batch boundary
    const int bb_b = gm / L_;         // per-row batch (hoisted out of k loop)
    const int ll   = gm % L_;

    // ---- compute role (ISA wave32 WMMA layouts) ----
    const int n   = lane & 15;        // N column / A row within M-tile
    const int hi  = lane >> 4;        // half-wave select
    const int ab  = hi * 8;           // A-frag K base: lanes<16 -> {0..7,16..23}
    const int bkb = hi * 8;           // B-frag K base: lanes<16 -> {0..7,16..23}
    const int o0  = wave * 64;

    // C init = bias (depends only on N -> broadcast over 8 M-rows per VGPR,
    // and identical for both M-tiles)
    v8f acc[2][4];
#pragma unroll
    for (int t = 0; t < 4; ++t) {
        const float bv = bias[o0 + t * 16 + n];
#pragma unroll
        for (int r = 0; r < 8; ++r) { acc[0][t][r] = bv; acc[1][t][r] = bv; }
    }

    for (int k = 0; k < K_; ++k) {
        // ---- gather + f32->f16 convert one 32x256 A tile into LDS ----
        const int j = (int)idx[(size_t)ll * K_ + k];
        const float4* src =
            (const float4*)(x + ((size_t)bb_b * L_ + j) * I_ + sub * 64);
        _Float16* dst = &As[row * STR + sub * 64];
#pragma unroll
        for (int q = 0; q < 8; ++q) {
            const float4 va = src[2 * q];
            const float4 vb = src[2 * q + 1];
            v8h h = {(_Float16)va.x, (_Float16)va.y, (_Float16)va.z, (_Float16)va.w,
                     (_Float16)vb.x, (_Float16)vb.y, (_Float16)vb.z, (_Float16)vb.w};
            *(v8h*)(dst + 8 * q) = h;
        }
        __syncthreads();

        // ---- 8 K-chunks of 32 over this mask slice ----
#pragma unroll 2
        for (int c = 0; c < 8; ++c) {
            // Two A fragments (one per 16-row M-tile); lane holds row n.
            v16h afrag[2];
#pragma unroll
            for (int mt = 0; mt < 2; ++mt) {
                const _Float16* arow = &As[(mt * 16 + n) * STR + c * 32 + ab];
                const v8h alo = *(const v8h*)arow;          // K = ab+0 .. ab+7
                const v8h ahi = *(const v8h*)(arow + 16);   // K = ab+16 .. ab+23
                afrag[mt] = __builtin_shufflevector(
                    alo, ahi, 0, 1, 2, 3, 4, 5, 6, 7, 8, 9, 10, 11, 12, 13, 14, 15);
            }
#pragma unroll
            for (int t = 0; t < 4; ++t) {
                // B fragment: lane = column o0+t*16+n, i-values c*32+bkb+{0..7,16..23}
                const _Float16* bp =
                    maskT + ((size_t)k * O_ + o0 + t * 16 + n) * I_ + c * 32 + bkb;
                const v16h bf = __builtin_shufflevector(
                    *(const v8h*)bp, *(const v8h*)(bp + 16),
                    0, 1, 2, 3, 4, 5, 6, 7, 8, 9, 10, 11, 12, 13, 14, 15);
                acc[0][t] = __builtin_amdgcn_wmma_f32_16x16x32_f16(
                    false, afrag[0], false, bf, (short)0, acc[0][t], false, false);
                acc[1][t] = __builtin_amdgcn_wmma_f32_16x16x32_f16(
                    false, afrag[1], false, bf, (short)0, acc[1][t], false, false);
            }
        }
        __syncthreads();   // A tile consumed before next k overwrites it
    }

    // ---- store: VGPR r of acc[mt][t] -> row m0 + mt*16 + r + hi*8 ----
#pragma unroll
    for (int mt = 0; mt < 2; ++mt) {
#pragma unroll
        for (int t = 0; t < 4; ++t) {
            const int o = o0 + t * 16 + n;
#pragma unroll
            for (int r = 0; r < 8; ++r) {
                out[((size_t)(m0 + mt * 16 + r + hi * 8)) * O_ + o] = acc[mt][t][r];
            }
        }
    }
}

// ---------------------------------------------------------------------------
// Inputs (setup_inputs order): x f32[B,L,I], index_tensor i64[L,K],
// mask f32[K,I,O], bias f32[O]. Output: f32[B,L,O].
// Workspace: 2 MB for maskT (f16 [K][O][I]).
// ---------------------------------------------------------------------------
extern "C" void kernel_launch(void* const* d_in, const int* in_sizes, int n_in,
                              void* d_out, int out_size, void* d_ws, size_t ws_size,
                              hipStream_t stream) {
    const float*     x    = (const float*)d_in[0];
    const long long* idx  = (const long long*)d_in[1];
    const float*     mask = (const float*)d_in[2];
    const float*     bias = (const float*)d_in[3];
    float*           out  = (float*)d_out;
    _Float16*        maskT = (_Float16*)d_ws;   // K_*O_*I_*2 = 2 MB

    dim3 ttb(32, 8);
    dim3 ttg(O_ / 32, I_ / 32, K_);
    transpose_mask_kernel<<<ttg, ttb, 0, stream>>>(mask, maskT);

    const int tiles = (B_ * L_) / 32;   // 1250
    iconv_wmma_kernel<<<tiles, 128, 0, stream>>>(x, idx, maskT, bias, out);
}
